// GateNet_31224412242353
// MI455X (gfx1250) — matrix operand order repaired and checked by reading, not compiled, not run on hardware
//
#include <hip/hip_runtime.h>
#include <hip/hip_bf16.h>
#include <math.h>

typedef __attribute__((ext_vector_type(16))) _Float16 v16h;
typedef __attribute__((ext_vector_type(8)))  float    v8f;
typedef __attribute__((__vector_size__(4 * sizeof(int)))) int vi4;

#define BN_EPS 1e-3f

// Async global->LDS path (gfx1250): guarded so either toolchain still compiles.
#if defined(__has_builtin)
#if __has_builtin(__builtin_amdgcn_global_load_async_to_lds_b128) && \
    __has_builtin(__builtin_amdgcn_s_wait_asynccnt)
#define USE_ASYNC_LDS 1
#endif
#endif
#ifndef USE_ASYNC_LDS
#define USE_ASYNC_LDS 0
#endif

#define GLOBAL_AS __attribute__((address_space(1)))
#define LDS_AS    __attribute__((address_space(3)))

// ---------------------------------------------------------------------------
// Weight repack: (3,3,3,Cin,Cout) f32 -> [chunk q][n][k=0..31] f16
// packed element kk = q*32 + k maps to flat weight index kk*Cout + n
// (since (((kd*3+kh)*3+kw)*Cin + c) == kk). Zero-pad kk >= 27*Cin.
// ---------------------------------------------------------------------------
__global__ void pack_weights_kernel(const float* __restrict__ w,
                                    _Float16* __restrict__ pw,
                                    int K, int Kpad, int Cout) {
    int idx = blockIdx.x * 256 + threadIdx.x;           // over (Kpad/32)*Cout*32
    int total = (Kpad >> 5) * Cout * 32;
    if (idx >= total) return;
    int k  = idx & 31;
    int qn = idx >> 5;
    int n  = qn % Cout;
    int q  = qn / Cout;
    int kk = q * 32 + k;
    pw[idx] = (kk < K) ? (_Float16)w[(size_t)kk * Cout + n] : (_Float16)0.0f;
}

// Precompute fused BN epilogue: y = scale*acc + shift; scale = g*rsqrt(v+eps),
// shift = scale*(bias - mean) + beta.
__global__ void bn_prep_kernel(const float* __restrict__ b, const float* __restrict__ g,
                               const float* __restrict__ be, const float* __restrict__ m,
                               const float* __restrict__ v,
                               float* __restrict__ scale, float* __restrict__ shift, int C) {
    int c = blockIdx.x * 64 + threadIdx.x;
    if (c < C) {
        float s = g[c] * rsqrtf(v[c] + BN_EPS);
        scale[c] = s;
        shift[c] = s * (b[c] - m[c]) + be[c];
    }
}

// ---------------------------------------------------------------------------
// Implicit-GEMM 3x3x3 stride-2 SAME conv + fused BN + ReLU, f16 WMMA.
//  block = 128 threads (4 waves). Wave w owns M-tile (blockIdx.x*4+w)*16 and
//  computes NT consecutive 16-col N tiles (A fragment reused across NT WMMAs).
//  The 16*NT x 32 B tile is staged in LDS, shared by all 4 waves, and
//  double-buffered via async global->LDS loads when available.
// ---------------------------------------------------------------------------
template <int CIN, int COUT, int DIN, int NT>
__global__ __launch_bounds__(128)
void conv_wmma_kernel(const void* __restrict__ inAct,          // f32 if CIN==1 else f16
                      const _Float16* __restrict__ pw,         // packed weights
                      const float* __restrict__ scale,
                      const float* __restrict__ shift,
                      _Float16* __restrict__ outAct) {
    constexpr int O     = DIN / 2;
    constexpr int K     = 27 * CIN;
    constexpr int NCH   = (K + 31) / 32;                 // K chunks of 32
    constexpr int CHPB  = (CIN >= 32) ? (CIN / 32) : 1;  // chunks per tap
    constexpr int TILE4 = NT * 64;                       // uint4 per B tile (NT*16 rows x 64B)

    __shared__ __align__(16) uint4 ldsB4[2][TILE4];

    const int tid   = threadIdx.x;
    const int lane  = tid & 31;
    const int wave  = tid >> 5;
    const int mBase = (blockIdx.x * 4 + wave) * 16;
    const int nBase = blockIdx.y * (16 * NT);
    const bool hiHalf = (lane >= 16);
    const uint4* gsrc4 = (const uint4*)pw;

    // Per-lane output coordinates (row m = lane & 15), fixed for whole kernel.
    int gm = mBase + (lane & 15);
    int ow = gm % O;  int t1 = gm / O;
    int oh = t1 % O;  int t2 = t1 / O;
    int od = t2 % O;  int bb = t2 / O;
    const int iw0 = ow * 2, ih0 = oh * 2, id0 = od * 2;  // SAME, stride 2: pad_lo = 0

    v8f acc[NT];
    #pragma unroll
    for (int j = 0; j < NT; ++j) acc[j] = (v8f){};

#if USE_ASYNC_LDS
    // Prologue: stage chunk 0 into buffer 0 asynchronously.
    for (int i = tid; i < TILE4; i += 128) {
        const uint4* g = gsrc4 + (size_t)(0 * COUT + nBase + (i >> 2)) * 4 + (i & 3);
        __builtin_amdgcn_global_load_async_to_lds_b128(
            (GLOBAL_AS vi4*)g, (LDS_AS vi4*)&ldsB4[0][i], 0, 0);
    }
#endif

    for (int q = 0; q < NCH; ++q) {
        const int buf = q & 1;
#if USE_ASYNC_LDS
        __builtin_amdgcn_s_wait_asynccnt(0);   // my portion of this tile landed
        __syncthreads();                        // everyone's portion landed; prev reads done
        if (q + 1 < NCH) {
            for (int i = tid; i < TILE4; i += 128) {
                const uint4* g = gsrc4 + (size_t)((q + 1) * COUT + nBase + (i >> 2)) * 4 + (i & 3);
                __builtin_amdgcn_global_load_async_to_lds_b128(
                    (GLOBAL_AS vi4*)g, (LDS_AS vi4*)&ldsB4[buf ^ 1][i], 0, 0);
            }
        }
#else
        __syncthreads();
        for (int i = tid; i < TILE4; i += 128)
            ldsB4[buf][i] = gsrc4[(size_t)(q * COUT + nBase + (i >> 2)) * 4 + (i & 3)];
        __syncthreads();
#endif

        // ---- A fragment (shared across NT WMMAs) ----
        v16h afrag = {};
        if constexpr (CIN == 1) {
            // Block 0: each K element is its own tap (kk = tap index, pad 27..31).
            const float* x = (const float*)inAct;
            const int off = hiHalf ? 8 : 0;
            #pragma unroll
            for (int j = 0; j < 16; ++j) {
                int kk = off + j + ((j >= 8) ? 8 : 0);
                _Float16 val = (_Float16)0.0f;
                if (kk < 27) {
                    int kd = kk / 9, r = kk % 9, kh = r / 3, kw = r % 3;
                    int id = id0 + kd, ih = ih0 + kh, iw = iw0 + kw;
                    if (id < DIN && ih < DIN && iw < DIN)
                        val = (_Float16)x[((size_t)(bb * DIN + id) * DIN + ih) * DIN + iw];
                }
                afrag[j] = val;
            }
        } else {
            // One tap per CHPB chunks; 32 contiguous channels per chunk.
            const _Float16* act = (const _Float16*)inAct;
            int t  = q / CHPB;
            int c0 = (q % CHPB) * 32;
            int kd = t / 9, r = t % 9, kh = r / 3, kw = r % 3;
            int id = id0 + kd, ih = ih0 + kh, iw = iw0 + kw;
            if (id < DIN && ih < DIN && iw < DIN) {
                const _Float16* p = act
                    + ((size_t)((bb * DIN + id) * DIN + ih) * DIN + iw) * CIN
                    + c0 + (hiHalf ? 8 : 0);
                *((uint4*)&afrag)     = *(const uint4*)p;         // K elems 0..7
                *((uint4*)&afrag + 1) = *(const uint4*)(p + 16);  // K elems 8..15
                __builtin_prefetch(p + CIN, 0, 1);                // next kw pixel
            }
        }

        // ---- NT B fragments from LDS, NT WMMAs reusing afrag ----
        const _Float16* lbase = (const _Float16*)ldsB4[buf];
        const int koff16 = hiHalf ? 8 : 0;
        #pragma unroll
        for (int j = 0; j < NT; ++j) {
            v16h bfrag;
            const _Float16* bp = lbase + (j * 16 + (lane & 15)) * 32 + koff16;
            *((uint4*)&bfrag)     = *(const uint4*)bp;
            *((uint4*)&bfrag + 1) = *(const uint4*)(bp + 16);
            acc[j] = __builtin_amdgcn_wmma_f32_16x16x32_f16(
                /*neg_a=*/false, afrag, /*neg_b=*/false, bfrag,
                /*c_mod=*/(short)0, acc[j], /*reuse_a=*/false, /*reuse_b=*/false);
        }
    }

    // ---- fused BN + ReLU epilogue, store f16 ----
    const int mRow = mBase + (hiHalf ? 8 : 0);
    #pragma unroll
    for (int j = 0; j < NT; ++j) {
        const int n = nBase + j * 16 + (lane & 15);
        const float s  = scale[n];
        const float sh = shift[n];
        #pragma unroll
        for (int r = 0; r < 8; ++r) {
            float y = s * acc[j][r] + sh;
            y = y > 0.0f ? y : 0.0f;
            outAct[(size_t)(mRow + r) * COUT + n] = (_Float16)y;
        }
    }
}

// ---------------------------------------------------------------------------
// Global spatial mean over 6^3 = 216 positions: (16, 216, 256) f16 -> (16,256) f32
// ---------------------------------------------------------------------------
__global__ void mean_kernel(const _Float16* __restrict__ act, float* __restrict__ out) {
    int b = blockIdx.x, c = threadIdx.x;  // 256 threads
    float s = 0.0f;
    for (int p = 0; p < 216; ++p)
        s += (float)act[((size_t)b * 216 + p) * 256 + c];
    out[b * 256 + c] = s * (1.0f / 216.0f);
}

// ---------------------------------------------------------------------------
// Head: hidden = relu(bn(h @ W1 + b1)); logits = hidden @ W2 + b2;
// top-2 mask; softmax over 8 experts. One block per batch element.
// ---------------------------------------------------------------------------
__global__ void head_kernel(const float* __restrict__ hmean,
                            const float* __restrict__ w1, const float* __restrict__ w1b,
                            const float* __restrict__ wg, const float* __restrict__ wbe,
                            const float* __restrict__ wm, const float* __restrict__ wv,
                            const float* __restrict__ w2, const float* __restrict__ w2b,
                            float* __restrict__ out) {
    __shared__ float hv[256];
    __shared__ float hid[256];
    __shared__ float logits[8];
    int b = blockIdx.x, c = threadIdx.x;   // 256 threads

    hv[c] = hmean[b * 256 + c];
    __syncthreads();

    float acc = 0.0f;
    for (int i = 0; i < 256; ++i) acc += hv[i] * w1[i * 256 + c];
    float s = wg[c] * rsqrtf(wv[c] + BN_EPS);
    float y = s * (acc + w1b[c] - wm[c]) + wbe[c];
    hid[c] = y > 0.0f ? y : 0.0f;
    __syncthreads();

    if (c < 8) {
        float a = w2b[c];
        for (int i = 0; i < 256; ++i) a += hid[i] * w2[i * 8 + c];
        logits[c] = a;
    }
    __syncthreads();

    if (c == 0) {
        float m1 = -INFINITY, m2 = -INFINITY;
        for (int e = 0; e < 8; ++e) {
            float v = logits[e];
            if (v > m1) { m2 = m1; m1 = v; } else if (v > m2) { m2 = v; }
        }
        float ex[8], denom = 0.0f;
        for (int e = 0; e < 8; ++e) {
            float v = logits[e];
            if (v >= m2) { ex[e] = __expf(v - m1); denom += ex[e]; } else ex[e] = 0.0f;
        }
        float inv = 1.0f / denom;
        for (int e = 0; e < 8; ++e) out[b * 8 + e] = ex[e] * inv;
    }
}

// ---------------------------------------------------------------------------
extern "C" void kernel_launch(void* const* d_in, const int* in_sizes, int n_in,
                              void* d_out, int out_size, void* d_ws, size_t ws_size,
                              hipStream_t stream) {
    (void)in_sizes; (void)n_in; (void)out_size; (void)ws_size;

    const float* x = (const float*)d_in[0];
    // per-block params: base = 1 + 6*i : k, b, g, be, m, v
    const float* K_[4];  const float* Bb[4]; const float* G[4];
    const float* Be[4];  const float* Mu[4]; const float* Va[4];
    for (int i = 0; i < 4; ++i) {
        K_[i] = (const float*)d_in[1 + 6 * i + 0];
        Bb[i] = (const float*)d_in[1 + 6 * i + 1];
        G [i] = (const float*)d_in[1 + 6 * i + 2];
        Be[i] = (const float*)d_in[1 + 6 * i + 3];
        Mu[i] = (const float*)d_in[1 + 6 * i + 4];
        Va[i] = (const float*)d_in[1 + 6 * i + 5];
    }
    const float* w1k = (const float*)d_in[25];
    const float* w1b = (const float*)d_in[26];
    const float* wg  = (const float*)d_in[27];
    const float* wbe = (const float*)d_in[28];
    const float* wm  = (const float*)d_in[29];
    const float* wv  = (const float*)d_in[30];
    const float* w2k = (const float*)d_in[31];
    const float* w2b = (const float*)d_in[32];
    float* out = (float*)d_out;

    // ---- workspace layout ----
    const int COUT[4] = {32, 64, 128, 256};
    const int CIN_[4] = {1, 32, 64, 128};
    const int DIN_[4] = {96, 48, 24, 12};
    size_t off = 0;
    char* ws = (char*)d_ws;
    auto take = [&](size_t bytes) { char* p = ws + off; off = (off + bytes + 255) & ~(size_t)255; return p; };

    _Float16* act[4];
    size_t Mtot[4];
    for (int i = 0; i < 4; ++i) {
        int O = DIN_[i] / 2;
        Mtot[i] = (size_t)16 * O * O * O;
        act[i] = (_Float16*)take(Mtot[i] * COUT[i] * sizeof(_Float16));
    }
    _Float16* pw[4];
    int Kpad[4];
    for (int i = 0; i < 4; ++i) {
        int K = 27 * CIN_[i];
        Kpad[i] = ((K + 31) / 32) * 32;
        pw[i] = (_Float16*)take((size_t)Kpad[i] * COUT[i] * sizeof(_Float16));
    }
    float *sc[4], *sh[4];
    for (int i = 0; i < 4; ++i) {
        sc[i] = (float*)take(COUT[i] * sizeof(float));
        sh[i] = (float*)take(COUT[i] * sizeof(float));
    }
    float* hmean = (float*)take(16 * 256 * sizeof(float));

    // ---- weight repack + BN prep ----
    for (int i = 0; i < 4; ++i) {
        int total = Kpad[i] * COUT[i];
        pack_weights_kernel<<<(total + 255) / 256, 256, 0, stream>>>(
            K_[i], pw[i], 27 * CIN_[i], Kpad[i], COUT[i]);
        bn_prep_kernel<<<(COUT[i] + 63) / 64, 64, 0, stream>>>(
            Bb[i], G[i], Be[i], Mu[i], Va[i], sc[i], sh[i], COUT[i]);
    }

    // ---- conv blocks (WMMA implicit GEMM) ----
    // grid.x = Mtiles/4 (4 waves per block), grid.y = Cout/(16*NT)
    conv_wmma_kernel<1,   32, 96, 2><<<dim3((unsigned)(Mtot[0] / 64), 1), 128, 0, stream>>>(
        (const void*)x, pw[0], sc[0], sh[0], act[0]);
    conv_wmma_kernel<32,  64, 48, 4><<<dim3((unsigned)(Mtot[1] / 64), 1), 128, 0, stream>>>(
        (const void*)act[0], pw[1], sc[1], sh[1], act[1]);
    conv_wmma_kernel<64, 128, 24, 4><<<dim3((unsigned)(Mtot[2] / 64), 2), 128, 0, stream>>>(
        (const void*)act[1], pw[2], sc[2], sh[2], act[2]);
    conv_wmma_kernel<128,256, 12, 4><<<dim3((unsigned)(Mtot[3] / 64), 4), 128, 0, stream>>>(
        (const void*)act[2], pw[3], sc[3], sh[3], act[3]);

    // ---- global mean + gating head ----
    mean_kernel<<<16, 256, 0, stream>>>(act[3], hmean);
    head_kernel<<<16, 256, 0, stream>>>(hmean, w1k, w1b, wg, wbe, wm, wv, w2k, w2b, out);
}
